// CompletionModel_44263932952636
// MI455X (gfx1250) — matrix operand order repaired and checked by reading, not compiled, not run on hardware
//
#include <hip/hip_runtime.h>
#include <cstdint>
#include <cstddef>

#define VOCAB 50000
#define DIM   300
#define DP    320      // DIM padded to multiple of 32 (10 k-steps)
#define HID   512
#define IMGD  2048
#define BATCH 256
#define SEQ   64
#define G3H   1536     // 3*HID
#define FEATD 3072     // 2*HID + IMGD
#define NTOP  1000

typedef __attribute__((ext_vector_type(16))) _Float16 v16h;
typedef __attribute__((ext_vector_type(8)))  float    v8f;

union FragU { v16h v; uint32_t u[8]; _Float16 h[16]; };

__device__ __forceinline__ v8f wmma_f16(v16h a, v16h b, v8f c) {
  // 8 args: (neg_a, A, neg_b, B, c_mod, C, reuse_a, reuse_b)
  return __builtin_amdgcn_wmma_f32_16x16x32_f16(false, a, false, b, (short)0, c,
                                                false, false);
}

// ---- fragment loaders (16-bit operands, contiguous k = 16*half + s) --------------
// A: 16x32 (MxK). lane = 16*half + m. VGPR v -> k0 = 2v + 16*half.
// Pairs contiguous -> dword loads
// (compiler merges these into global_load_b128 pairs).
__device__ __forceinline__ v16h load_a_row(const _Float16* rowp, int kb, int lane) {
  int hs = (lane >> 4) & 1;
  const _Float16* p = rowp + kb + 16 * hs;
  FragU f;
#pragma unroll
  for (int v = 0; v < 8; ++v) {
    int k0 = 2 * v;
    f.u[v] = *(const uint32_t*)(p + k0);
  }
  return f.v;
}

// B: 32x16 (KxN) sourced from row-major W[n, k] (ld = K stride).
// lane = 16*half + n. VGPR v -> k0 = 2v + 16*half. Pairs contiguous in k.
__device__ __forceinline__ v16h load_b(const _Float16* W, int ld, int nrow, int kb, int lane) {
  int hs = (lane >> 4) & 1;
  const _Float16* p = W + (size_t)nrow * ld + kb + 16 * hs;
  FragU f;
#pragma unroll
  for (int v = 0; v < 8; ++v) f.u[v] = *(const uint32_t*)(p + 2 * v);
  return f.v;
}

__device__ __forceinline__ v16h load_b_guard(const _Float16* W, int ld, int nrow, int nmax,
                                             int kb, int lane) {
  FragU f;
  if (nrow >= nmax) {
#pragma unroll
    for (int v = 0; v < 8; ++v) f.u[v] = 0u;
    return f.v;
  }
  int hs = (lane >> 4) & 1;
  const _Float16* p = W + (size_t)nrow * ld + kb + 16 * hs;
#pragma unroll
  for (int v = 0; v < 8; ++v) f.u[v] = *(const uint32_t*)(p + 2 * v);
  return f.v;
}

// ---- utility kernels ------------------------------------------------------
__global__ void __launch_bounds__(256) k_convert_pad(const float* in, _Float16* out,
                                                     int rows, int cin, int cout) {
  size_t n = (size_t)rows * cout;
  for (size_t i = (size_t)blockIdx.x * blockDim.x + threadIdx.x; i < n;
       i += (size_t)gridDim.x * blockDim.x) {
    int r = (int)(i / cout), c = (int)(i % cout);
    out[i] = (c < cin) ? (_Float16)in[(size_t)r * cin + c] : (_Float16)0.f;
  }
}

__global__ void __launch_bounds__(256) k_zero_u32(uint32_t* p, size_t n) {
  for (size_t i = (size_t)blockIdx.x * blockDim.x + threadIdx.x; i < n;
       i += (size_t)gridDim.x * blockDim.x) p[i] = 0u;
}

// 1/(||row||+eps), one wave per row
__global__ void __launch_bounds__(256) k_rownorm(const float* in, float* out,
                                                 int rows, int cols) {
  int wid  = (int)((blockIdx.x * blockDim.x + threadIdx.x) >> 5);
  int lane = threadIdx.x & 31;
  if (wid >= rows) return;
  const float* row = in + (size_t)wid * cols;
  float s = 0.f;
  for (int c = lane; c < cols; c += 32) { float v = row[c]; s += v * v; }
#pragma unroll
  for (int off = 16; off; off >>= 1) s += __shfl_xor(s, off, 32);
  if (lane == 0) out[wid] = 1.0f / (sqrtf(s) + 1e-6f);
}

// ---- xg = embedding[sent] @ Wx^T + bx  : [16384, 1536] f32 ---------------
__global__ void __launch_bounds__(256) k_xg_gemm(const _Float16* emb16, const int* sent,
                                                 const _Float16* Wx16, const float* bx,
                                                 float* xg) {
  int wave = (int)((blockIdx.x * blockDim.x + threadIdx.x) >> 5);
  int lane = threadIdx.x & 31;
  int nt = wave % (G3H / 16);          // 96 N tiles
  int mt = wave / (G3H / 16);          // 1024 M tiles
  if (mt >= (BATCH * SEQ) / 16) return;
  int hs = lane >> 4, m = lane & 15;
  int row = mt * 16 + m;
  int tok = sent[row];                 // A-side gather
  const _Float16* arow = emb16 + (size_t)tok * DP;
  v8f acc = {};
  for (int kb = 0; kb < DP; kb += 32) {
    v16h a = load_a_row(arow, kb, lane);
    v16h b = load_b(Wx16, DP, nt * 16 + m /*n==lane&15*/, kb, lane);
    acc = wmma_f16(a, b, acc);
  }
  int col = nt * 16 + (lane & 15);
  float bias = bx[col];
#pragma unroll
  for (int r = 0; r < 8; ++r) {
    int mm = mt * 16 + r + 8 * hs;
    xg[(size_t)mm * G3H + col] = acc[r] + bias;
  }
}

// ---- persistent GRU: all 64 time steps in one launch ---------------------
// Grid must be exactly 64 blocks x 256 threads (512 waves = 16 x 32 tiles);
// no early returns (grid-wide barrier). Step 0 exploits h0 == 0 (hg = bh),
// skipping the K-loop entirely (wave-uniform branch, EXEC stays all-ones).
__global__ void __launch_bounds__(256) k_gru_persistent(
    const float* xg, const _Float16* Wh16, const float* bh,
    float* h32a, float* h32b, _Float16* h16a, _Float16* h16b,
    const int* lens, float* feat, unsigned* synccnt) {
  int wave = (int)((blockIdx.x * blockDim.x + threadIdx.x) >> 5);
  int lane = threadIdx.x & 31;
  int tid  = threadIdx.x;
  int nt = wave & 31;                  // 32 H tiles
  int mt = wave >> 5;                  // 16 batch tiles (exact: 512 waves)
  int hs = lane >> 4, ml = lane & 15;
  int j  = nt * 16 + (lane & 15);      // H column for this lane's outputs
  float bhr = bh[j], bhz = bh[HID + j], bhn = bh[2 * HID + j];
  int mrow = mt * 16 + ml;

  for (int s = 0; s < SEQ; ++s) {
    const float*    hp   = (s & 1) ? h32b : h32a;
    const _Float16* hp16 = (s & 1) ? h16b : h16a;
    float*          hn   = (s & 1) ? h32a : h32b;
    _Float16*       hn16 = (s & 1) ? h16a : h16b;

    v8f ar = {}, az = {}, an = {};
    if (s > 0) {                       // s == 0: h0 == 0 -> hg = bh
      const _Float16* arow = hp16 + (size_t)mrow * HID;
      for (int kb = 0; kb < HID; kb += 32) {
        v16h a  = load_a_row(arow, kb, lane);
        v16h br = load_b(Wh16, HID, 0 * HID + nt * 16 + ml, kb, lane);
        v16h bz = load_b(Wh16, HID, 1 * HID + nt * 16 + ml, kb, lane);
        v16h bn = load_b(Wh16, HID, 2 * HID + nt * 16 + ml, kb, lane);
        ar = wmma_f16(a, br, ar);
        az = wmma_f16(a, bz, az);
        an = wmma_f16(a, bn, an);
      }
    }
#pragma unroll
    for (int r = 0; r < 8; ++r) {
      int b = mt * 16 + r + 8 * hs;
      size_t xbase = ((size_t)b * SEQ + s) * G3H;
      float xr = xg[xbase + j];
      float xz = xg[xbase + HID + j];
      float xn = xg[xbase + 2 * HID + j];
      float hpv = (s > 0) ? hp[(size_t)b * HID + j] : 0.f;
      float rg = 1.f / (1.f + __expf(-(xr + ar[r] + bhr)));
      float zg = 1.f / (1.f + __expf(-(xz + az[r] + bhz)));
      float ng = tanhf(xn + rg * (an[r] + bhn));
      float hv = (1.f - zg) * ng + zg * hpv;
      hn[(size_t)b * HID + j]   = hv;
      hn16[(size_t)b * HID + j] = (_Float16)hv;
      if (lens[b] - 1 == s) feat[(size_t)b * HID + j] = hv;
    }

    // grid-wide barrier: monotonic counter, release on arrive / acquire spin
    __syncthreads();
    if (tid == 0) {
      __hip_atomic_fetch_add(synccnt, 1u, __ATOMIC_RELEASE, __HIP_MEMORY_SCOPE_AGENT);
      unsigned target = (unsigned)gridDim.x * (unsigned)(s + 1);
      while (__hip_atomic_load(synccnt, __ATOMIC_ACQUIRE, __HIP_MEMORY_SCOPE_AGENT) < target)
        __builtin_amdgcn_s_sleep(1);
    }
    __syncthreads();
  }
}

// ---- feature = concat(feat_f, feat_b, image) -> f16 ----------------------
__global__ void __launch_bounds__(256) k_feature(const float* ff, const float* fb,
                                                 const float* image, _Float16* out) {
  size_t n = (size_t)BATCH * FEATD;
  for (size_t i = (size_t)blockIdx.x * blockDim.x + threadIdx.x; i < n;
       i += (size_t)gridDim.x * blockDim.x) {
    int b = (int)(i / FEATD), c = (int)(i % FEATD);
    float v = (c < HID) ? ff[(size_t)b * HID + c]
            : (c < 2 * HID) ? fb[(size_t)b * HID + (c - HID)]
            : image[(size_t)b * IMGD + (c - 2 * HID)];
    out[i] = (_Float16)v;
  }
}

// ---- predict = feature @ W_pred^T + b_pred : [256, 300] ------------------
__global__ void __launch_bounds__(256) k_predict_gemm(const _Float16* feat16,
                                                      const _Float16* Wp16,
                                                      const float* bpred,
                                                      float* out, _Float16* pred16) {
  int wave = (int)((blockIdx.x * blockDim.x + threadIdx.x) >> 5);
  int lane = threadIdx.x & 31;
  int nt = wave % 19;                  // 19 N tiles (304 cols, 300 valid)
  int mt = wave / 19;                  // 16 batch tiles
  if (mt >= BATCH / 16) return;
  int hs = lane >> 4, m = lane & 15;
  const _Float16* arow = feat16 + (size_t)(mt * 16 + m) * FEATD;
  v8f acc = {};
  for (int kb = 0; kb < FEATD; kb += 32) {
    v16h a = load_a_row(arow, kb, lane);
    v16h b = load_b_guard(Wp16, FEATD, nt * 16 + m, DIM, kb, lane);
    acc = wmma_f16(a, b, acc);
  }
  int col = nt * 16 + (lane & 15);
#pragma unroll
  for (int r = 0; r < 8; ++r) {
    int b = mt * 16 + r + 8 * hs;
    if (col < DIM) {
      float v = acc[r] + bpred[col];
      out[(size_t)b * DIM + col]   = v;           // d_out predict region
      pred16[(size_t)b * DP + col] = (_Float16)v;
    }
  }
}

// ---- sim = normalize(predict) @ normalize(emb)^T : [256, 50000] ----------
__global__ void __launch_bounds__(256) k_sim_gemm(const _Float16* pred16,
                                                  const _Float16* emb16,
                                                  const float* pnorm, const float* enorm,
                                                  float* sim) {
  int wave = (int)((blockIdx.x * blockDim.x + threadIdx.x) >> 5);
  int lane = threadIdx.x & 31;
  int nt = wave % (VOCAB / 16);        // 3125 N tiles (exact)
  int mt = wave / (VOCAB / 16);        // 16 batch tiles
  if (mt >= BATCH / 16) return;
  int hs = lane >> 4, m = lane & 15;
  const _Float16* arow = pred16 + (size_t)(mt * 16 + m) * DP;
  v8f acc = {};
  for (int kb = 0; kb < DP; kb += 32) {
    v16h a = load_a_row(arow, kb, lane);
    v16h b = load_b(emb16, DP, nt * 16 + m, kb, lane);
    acc = wmma_f16(a, b, acc);
  }
  int col = nt * 16 + (lane & 15);
  float en = enorm[col];
#pragma unroll
  for (int r = 0; r < 8; ++r) {
    int b = mt * 16 + r + 8 * hs;
    sim[(size_t)b * VOCAB + col] = acc[r] * pnorm[b] * en;
  }
}

// ---- top-1000 of 50000 per row: radix-select + bitonic sort --------------
__device__ __forceinline__ unsigned sortkey(float v) {
  unsigned u = __float_as_uint(v);
  return (u >> 31) ? ~u : (u | 0x80000000u);
}

// Ordered (deterministic) stream-compaction of elements matching a bin
// predicate, in index order, capped at `limit`, into keys[start..).
__device__ __forceinline__ void collect_pass(const float* row, int t, bool eq,
                                             int start, int limit,
                                             unsigned long long* keys, int* wave_tot,
                                             int tid) {
  int lane = tid & 31, wid = tid >> 5;
  int base = 0;
  for (int chunk = 0; chunk < VOCAB && base < limit; chunk += 256) {
    int i = chunk + tid;
    bool pred = false; unsigned key = 0;
    if (i < VOCAB) {
      key = sortkey(row[i]);
      int bin = (int)(key >> 21);
      pred = eq ? (bin == t) : (bin > t);
    }
    unsigned mask32 = (unsigned)__ballot(pred);     // wave32: low 32 bits
    int lp = __popc(mask32 & ((1u << lane) - 1u));
    if (lane == 0) wave_tot[wid] = __popc(mask32);
    __syncthreads();
    int wp = 0, tot = 0;
#pragma unroll
    for (int w = 0; w < 8; ++w) { if (w < wid) wp += wave_tot[w]; tot += wave_tot[w]; }
    int pos = base + wp + lp;
    if (pred && pos < limit)
      keys[start + pos] = (((unsigned long long)(~key)) << 32) | (unsigned)i;
    __syncthreads();                                 // before next wave_tot write
    base += tot;                                     // uniform across block
  }
  __syncthreads();
}

__global__ void __launch_bounds__(256) k_topk(const float* sim, const int* label,
                                              float* out_topk, int* labelpos) {
  __shared__ int hist[2048];
  __shared__ unsigned long long keys[1024];
  __shared__ int wave_tot[8];
  __shared__ int sh_t, sh_above, sh_lp;
  int b = blockIdx.x, tid = threadIdx.x;
  const float* row = sim + (size_t)b * VOCAB;

  for (int i = tid; i < 2048; i += 256) hist[i] = 0;
  __syncthreads();
  for (int i = tid; i < VOCAB; i += 256)
    atomicAdd(&hist[sortkey(row[i]) >> 21], 1);
  __syncthreads();
  if (tid == 0) {
    int cum = 0, t = 0, above = 0;
    for (int bin = 2047; bin >= 0; --bin) {
      cum += hist[bin];
      if (cum >= NTOP) { t = bin; above = cum - hist[bin]; break; }
    }
    sh_t = t; sh_above = above; sh_lp = 0x7FFFFFFF;
  }
  __syncthreads();
  int t = sh_t, above = sh_above, need = NTOP - above;

  collect_pass(row, t, false, 0,     above, keys, wave_tot, tid);  // bins > t
  collect_pass(row, t, true,  above, need,  keys, wave_tot, tid);  // bin == t
  for (int i = NTOP + tid; i < 1024; i += 256) keys[i] = ~0ULL;    // pad
  __syncthreads();

  // bitonic ascending on key64 = (~sortkey << 32) | idx -> values descending,
  // ties broken by smaller index first.
  for (int ksz = 2; ksz <= 1024; ksz <<= 1) {
    for (int j = ksz >> 1; j > 0; j >>= 1) {
      for (int i = tid; i < 1024; i += 256) {
        int ixj = i ^ j;
        if (ixj > i) {
          bool asc = ((i & ksz) == 0);
          unsigned long long a = keys[i], c = keys[ixj];
          if ((a > c) == asc) { keys[i] = c; keys[ixj] = a; }
        }
      }
      __syncthreads();
    }
  }

  int lab = label[b];
  for (int i = tid; i < NTOP; i += 256) {
    int idx = (int)(unsigned)(keys[i] & 0xFFFFFFFFULL);
    out_topk[(size_t)b * NTOP + i] = (float)idx;
    if (idx == lab) atomicMin(&sh_lp, i);
  }
  __syncthreads();
  if (tid == 0) labelpos[b] = sh_lp;
}

__global__ void __launch_bounds__(256) k_metrics(const int* labelpos, float* out) {
  __shared__ int cnt[4];
  if (threadIdx.x < 4) cnt[threadIdx.x] = 0;
  __syncthreads();
  int p = labelpos[threadIdx.x];
  if (p < 1)    atomicAdd(&cnt[0], 1);
  if (p < 10)   atomicAdd(&cnt[1], 1);
  if (p < 100)  atomicAdd(&cnt[2], 1);
  if (p < NTOP) atomicAdd(&cnt[3], 1);
  __syncthreads();
  if (threadIdx.x < 4) out[threadIdx.x] = (float)cnt[threadIdx.x] / (float)BATCH;
}

// ---- host orchestration ---------------------------------------------------
extern "C" void kernel_launch(void* const* d_in, const int* in_sizes, int n_in,
                              void* d_out, int out_size, void* d_ws, size_t ws_size,
                              hipStream_t stream) {
  (void)in_sizes; (void)n_in; (void)out_size; (void)ws_size;
  const int*   sent_f = (const int*)  d_in[0];
  const int*   len_f  = (const int*)  d_in[1];
  const int*   sent_b = (const int*)  d_in[2];
  const int*   len_b  = (const int*)  d_in[3];
  const float* image  = (const float*)d_in[4];
  const int*   label  = (const int*)  d_in[5];
  const float* emb    = (const float*)d_in[6];
  const float* Wx_f   = (const float*)d_in[7];
  const float* Wh_f   = (const float*)d_in[8];
  const float* bx_f   = (const float*)d_in[9];
  const float* bh_f   = (const float*)d_in[10];
  const float* Wx_b   = (const float*)d_in[11];
  const float* Wh_b   = (const float*)d_in[12];
  const float* bx_b   = (const float*)d_in[13];
  const float* bh_b   = (const float*)d_in[14];
  const float* W_pred = (const float*)d_in[15];
  const float* b_pred = (const float*)d_in[16];

  float* out         = (float*)d_out;
  float* out_predict = out;                               // [256, 300]
  float* out_topk    = out + BATCH * DIM;                 // [256, 1000]
  float* out_metrics = out + BATCH * DIM + BATCH * NTOP;  // [4]

  // workspace carving (256B aligned)
  char* w = (char*)d_ws;
  auto take = [&](size_t bytes) -> void* {
    void* p = (void*)w; w += (bytes + 255) & ~(size_t)255; return p;
  };
  _Float16* emb16   = (_Float16*)take((size_t)VOCAB * DP * 2);
  _Float16* Wx16f   = (_Float16*)take((size_t)G3H * DP * 2);
  _Float16* Wx16b   = (_Float16*)take((size_t)G3H * DP * 2);
  _Float16* Wh16f   = (_Float16*)take((size_t)G3H * HID * 2);
  _Float16* Wh16b   = (_Float16*)take((size_t)G3H * HID * 2);
  _Float16* Wp16    = (_Float16*)take((size_t)DIM * FEATD * 2);
  float*    xg      = (float*)   take((size_t)BATCH * SEQ * G3H * 4);  // reused per dir
  float*    h32a    = (float*)   take((size_t)BATCH * HID * 4);
  float*    h32b    = (float*)   take((size_t)BATCH * HID * 4);
  _Float16* h16a    = (_Float16*)take((size_t)BATCH * HID * 2);
  _Float16* h16b    = (_Float16*)take((size_t)BATCH * HID * 2);
  float*    featf   = (float*)   take((size_t)BATCH * HID * 4);
  float*    featb   = (float*)   take((size_t)BATCH * HID * 4);
  _Float16* feat16  = (_Float16*)take((size_t)BATCH * FEATD * 2);
  _Float16* pred16  = (_Float16*)take((size_t)BATCH * DP * 2);
  float*    pnorm   = (float*)   take((size_t)BATCH * 4);
  float*    enorm   = (float*)   take((size_t)VOCAB * 4);
  float*    sim     = (float*)   take((size_t)BATCH * VOCAB * 4);
  int*      lpos    = (int*)     take((size_t)BATCH * 4);
  unsigned* synccnt = (unsigned*)take(256);

  dim3 blk(256);
  auto cdiv = [](size_t a, size_t b) -> unsigned { return (unsigned)((a + b - 1) / b); };

  // f16 conversions (K padded to DP where K==DIM)
  k_convert_pad<<<cdiv((size_t)VOCAB * DP, 256), blk, 0, stream>>>(emb, emb16, VOCAB, DIM, DP);
  k_convert_pad<<<cdiv((size_t)G3H * DP, 256), blk, 0, stream>>>(Wx_f, Wx16f, G3H, DIM, DP);
  k_convert_pad<<<cdiv((size_t)G3H * DP, 256), blk, 0, stream>>>(Wx_b, Wx16b, G3H, DIM, DP);
  k_convert_pad<<<cdiv((size_t)G3H * HID, 256), blk, 0, stream>>>(Wh_f, Wh16f, G3H, HID, HID);
  k_convert_pad<<<cdiv((size_t)G3H * HID, 256), blk, 0, stream>>>(Wh_b, Wh16b, G3H, HID, HID);
  k_convert_pad<<<cdiv((size_t)DIM * FEATD, 256), blk, 0, stream>>>(W_pred, Wp16, DIM, FEATD, FEATD);
  k_rownorm<<<cdiv((size_t)VOCAB * 32, 256), blk, 0, stream>>>(emb, enorm, VOCAB, DIM);

  // two GRU directions (xg buffer reused; stream ordering serializes)
  const int*      sents[2] = { sent_f, sent_b };
  const int*      lens_[2] = { len_f, len_b };
  const _Float16* Wx16d[2] = { Wx16f, Wx16b };
  const _Float16* Wh16d[2] = { Wh16f, Wh16b };
  const float*    bxd[2]   = { bx_f, bx_b };
  const float*    bhd[2]   = { bh_f, bh_b };
  float*          featd[2] = { featf, featb };

  unsigned xg_blocks  = (unsigned)(((size_t)(BATCH * SEQ / 16) * (G3H / 16)) / 8); // 12288
  unsigned gru_blocks = (unsigned)(((size_t)(BATCH / 16) * (HID / 16)) / 8);       // 64

  for (int dir = 0; dir < 2; ++dir) {
    k_xg_gemm<<<xg_blocks, blk, 0, stream>>>(emb16, sents[dir], Wx16d[dir], bxd[dir], xg);
    k_zero_u32<<<cdiv((size_t)BATCH * HID, 256), blk, 0, stream>>>(
        (uint32_t*)featd[dir], (size_t)BATCH * HID);
    k_zero_u32<<<1, blk, 0, stream>>>((uint32_t*)synccnt, 1);   // reset grid barrier
    k_gru_persistent<<<gru_blocks, blk, 0, stream>>>(
        xg, Wh16d[dir], bhd[dir], h32a, h32b, h16a, h16b,
        lens_[dir], featd[dir], synccnt);
  }

  // fuse + predict + retrieval
  k_feature<<<cdiv((size_t)BATCH * FEATD, 256), blk, 0, stream>>>(featf, featb, image, feat16);
  k_zero_u32<<<cdiv((size_t)BATCH * DP / 2, 256), blk, 0, stream>>>(
      (uint32_t*)pred16, (size_t)BATCH * DP / 2);
  k_predict_gemm<<<(16 * 19) / 8, blk, 0, stream>>>(feat16, Wp16, b_pred, out_predict, pred16);
  k_rownorm<<<cdiv((size_t)BATCH * 32, 256), blk, 0, stream>>>(out_predict, pnorm, BATCH, DIM);
  k_sim_gemm<<<(unsigned)(((size_t)(BATCH / 16) * (VOCAB / 16)) / 8), blk, 0, stream>>>(
      pred16, emb16, pnorm, enorm, sim);

  // top-k + metrics
  k_topk<<<BATCH, blk, 0, stream>>>(sim, label, out_topk, lpos);
  k_metrics<<<1, blk, 0, stream>>>(lpos, out_metrics);
}
